// EdgeUpdate_85169201480454
// MI455X (gfx1250) — matrix operand order repaired and checked by reading, not compiled, not run on hardware
//
#include <hip/hip_runtime.h>
#include <hip/hip_fp16.h>

typedef __attribute__((ext_vector_type(16))) _Float16 v16h;
typedef __attribute__((ext_vector_type(8)))  _Float16 v8h;
typedef __attribute__((ext_vector_type(4)))  _Float16 v4h;
typedef __attribute__((ext_vector_type(8)))  float    v8f;

#define N_NODES 50000
#define N_EDGES 500000
#define D       128      // NODE_DIM == EDGE_DIM
#define KDIM    384      // 2*D + D  (concat of src, dst, edge_attr)
#define ODIM    256      // 2*D     (delta | gate_logit)
#define EPSLN   1e-5f

// ---------------------------------------------------------------------------
// Kernel 1: per-node LayerNorm, f32 -> f16 table (12.8 MB, L2-resident).
// One wave32 per node; lane l owns features 4l..4l+3.
// ---------------------------------------------------------------------------
__global__ __launch_bounds__(256) void node_ln_kernel(
    const float* __restrict__ x,
    const float* __restrict__ gamma,
    const float* __restrict__ beta,
    _Float16* __restrict__ xn)
{
    const int tid  = threadIdx.x;
    const int lane = tid & 31;
    const int node = blockIdx.x * 8 + (tid >> 5);

    const float4 v = ((const float4*)(x + (size_t)node * D))[lane];
    float s  = v.x + v.y + v.z + v.w;
    float sq = v.x * v.x + v.y * v.y + v.z * v.z + v.w * v.w;
#pragma unroll
    for (int m = 16; m >= 1; m >>= 1) {
        s  += __shfl_xor(s,  m, 32);
        sq += __shfl_xor(sq, m, 32);
    }
    const float mu  = s * (1.0f / D);
    const float var = sq * (1.0f / D) - mu * mu;
    const float inv = rsqrtf(var + EPSLN);

    const float4 g  = ((const float4*)gamma)[lane];
    const float4 be = ((const float4*)beta)[lane];
    v4h h;
    h.x = (_Float16)((v.x - mu) * inv * g.x + be.x);
    h.y = (_Float16)((v.y - mu) * inv * g.y + be.y);
    h.z = (_Float16)((v.z - mu) * inv * g.z + be.z);
    h.w = (_Float16)((v.w - mu) * inv * g.w + be.w);
    *(v4h*)(xn + (size_t)node * D + lane * 4) = h;
}

// ---------------------------------------------------------------------------
// Kernel 2: W [384][256] f32 -> WT [256][384] f16 so that a B-fragment
// (one N-column, two runs of 8 consecutive K halves) is two contiguous b128 loads.
// ---------------------------------------------------------------------------
__global__ __launch_bounds__(256) void wt_kernel(
    const float* __restrict__ W, _Float16* __restrict__ WT)
{
    const int idx = blockIdx.x * 256 + threadIdx.x;   // over ODIM*KDIM
    const int n = idx / KDIM;
    const int k = idx - n * KDIM;
    WT[idx] = (_Float16)W[(size_t)k * ODIM + n];
}

// ---------------------------------------------------------------------------
// Kernel 3: 32 edges per block, 8 waves. WMMA f32<=f16 16x16x32.
//   wave w: M-tile mt = w>>2 (16 edges), N-tiles (w&3)*4 .. +3 (64 cols)
//   K loop: 12 steps of 32  ->  48 v_wmma per wave.
// ---------------------------------------------------------------------------
__global__ __launch_bounds__(256) void edge_kernel(
    const int*      __restrict__ eidx,   // [2][E]
    const float*    __restrict__ ea,     // [E][128]
    const _Float16* __restrict__ xn,     // [N][128] f16 (normalized)
    const _Float16* __restrict__ WT,     // [256][384] f16
    const float*    __restrict__ bias,   // [256]
    const float*    __restrict__ egamma, // [128]
    const float*    __restrict__ ebeta,  // [128]
    float*          __restrict__ out)    // [E][128]
{
    __shared__ __align__(16) _Float16 lds_m[32 * KDIM];   // 24 KB A-panel
    __shared__ __align__(16) float    lds_acc[32 * ODIM]; // 32 KB raw GEMM out

    const int tid = threadIdx.x;
    const int e0  = blockIdx.x * 32;

    // ---- Stage A-panel: m[e] = [xn[src] | xn[dst] | f16(edge_attr)] --------
    // 32 rows * 384 halves = 1536 8-half vectors; 6 per thread.
#pragma unroll
    for (int it = 0; it < 6; ++it) {
        const int v  = it * 256 + tid;
        const int e  = v / 48;
        const int f8 = (v - e * 48) * 8;
        const int ge = e0 + e;
        v8h h;
        if (f8 < 128) {
            const int s = eidx[ge];
            h = *(const v8h*)(xn + (size_t)s * D + f8);
        } else if (f8 < 256) {
            const int dn = eidx[N_EDGES + ge];
            h = *(const v8h*)(xn + (size_t)dn * D + (f8 - 128));
        } else {
            const float* p = ea + (size_t)ge * D + (f8 - 256);
#pragma unroll
            for (int i = 0; i < 8; ++i) h[i] = (_Float16)p[i];
        }
        *(v8h*)(lds_m + e * KDIM + f8) = h;
    }
    __syncthreads();

    // ---- GEMM: acc[2 Mtiles][16 Ntiles] distributed over 8 waves ----------
    const int lane  = tid & 31;
    const int w     = tid >> 5;
    const int mt    = w >> 2;                 // 0..1
    const int ng    = w & 3;                  // N-tile group: cols ng*64..+63
    const int mrow  = mt * 16 + (lane & 15);  // A row for this lane
    const int koffA = (lane >= 16) ? 8  : 0;  // ISA 16-bit A 16x32 layout
    const int koffB = (lane >= 16) ? 8  : 0;  // ISA 16-bit B 32x16 layout
    const int ncol0 = ng * 64 + (lane & 15);

    v8f acc[4];
#pragma unroll
    for (int i = 0; i < 4; ++i)
#pragma unroll
        for (int r = 0; r < 8; ++r) acc[i][r] = 0.0f;

    for (int k0 = 0; k0 < KDIM; k0 += 32) {
        const _Float16* ap = lds_m + mrow * KDIM + k0 + koffA;
        const v8h a0 = *(const v8h*)(ap);        // K = k0+koffA   .. +7
        const v8h a1 = *(const v8h*)(ap + 16);   // K = k0+koffA+16.. +7
        const v16h A = __builtin_shufflevector(a0, a1,
            0,1,2,3,4,5,6,7,8,9,10,11,12,13,14,15);
#pragma unroll
        for (int i = 0; i < 4; ++i) {
            const int n = ncol0 + i * 16;
            const _Float16* bp = WT + (size_t)n * KDIM + k0 + koffB;
            const v8h b0 = *(const v8h*)(bp);
            const v8h b1 = *(const v8h*)(bp + 16);
            const v16h B = __builtin_shufflevector(b0, b1,
                0,1,2,3,4,5,6,7,8,9,10,11,12,13,14,15);
            acc[i] = __builtin_amdgcn_wmma_f32_16x16x32_f16(
                false, A, false, B, (short)0, acc[i], false, false);
        }
    }

    // ---- Spill accumulators to LDS (C/D layout: M = r + 8*(lane>=16)) -----
    {
        const int erow = mt * 16 + ((lane >= 16) ? 8 : 0);
        const int col  = lane & 15;
#pragma unroll
        for (int i = 0; i < 4; ++i) {
            const int n = (ng * 4 + i) * 16 + col;
#pragma unroll
            for (int r = 0; r < 8; ++r)
                lds_acc[(erow + r) * ODIM + n] = acc[i][r];
        }
    }
    __syncthreads();

    // ---- Epilogue 1: bias + ReLU, split, sigmoid gate, residual -----------
    // 32 edges * 128 cols = 4096 items; each owned by exactly one thread,
    // writes only touch cols < 128 while gate reads touch cols >= 128.
#pragma unroll
    for (int it = 0; it < 16; ++it) {
        const int v = it * 256 + tid;
        const int e = v >> 7;
        const int j = v & 127;
        float zd = lds_acc[e * ODIM + j]       + bias[j];
        float zg = lds_acc[e * ODIM + 128 + j] + bias[128 + j];
        zd = fmaxf(zd, 0.0f);
        zg = fmaxf(zg, 0.0f);
        const float gate = 1.0f / (1.0f + __expf(-zg));
        const float o = ea[(size_t)(e0 + e) * D + j] + gate * zd;
        lds_acc[e * ODIM + j] = o;
    }
    __syncthreads();

    // ---- Epilogue 2: per-edge LayerNorm (4 edges per wave) ----------------
#pragma unroll
    for (int q = 0; q < 4; ++q) {
        const int le = w * 4 + q;
        float vals[4];
        float s = 0.0f, sq = 0.0f;
#pragma unroll
        for (int p = 0; p < 4; ++p) {
            vals[p] = lds_acc[le * ODIM + lane + 32 * p];
            s  += vals[p];
            sq += vals[p] * vals[p];
        }
#pragma unroll
        for (int m = 16; m >= 1; m >>= 1) {
            s  += __shfl_xor(s,  m, 32);
            sq += __shfl_xor(sq, m, 32);
        }
        const float mu  = s * (1.0f / D);
        const float var = sq * (1.0f / D) - mu * mu;
        const float inv = rsqrtf(var + EPSLN);
#pragma unroll
        for (int p = 0; p < 4; ++p) {
            const int f = lane + 32 * p;
            out[(size_t)(e0 + le) * D + f] =
                (vals[p] - mu) * inv * egamma[f] + ebeta[f];
        }
    }
}

// ---------------------------------------------------------------------------
extern "C" void kernel_launch(void* const* d_in, const int* in_sizes, int n_in,
                              void* d_out, int out_size, void* d_ws, size_t ws_size,
                              hipStream_t stream)
{
    const float* x      = (const float*)d_in[0];
    const int*   eidx   = (const int*)  d_in[1];
    const float* ea     = (const float*)d_in[2];
    const float* W      = (const float*)d_in[3];
    const float* b      = (const float*)d_in[4];
    const float* ngamma = (const float*)d_in[5];
    const float* nbeta  = (const float*)d_in[6];
    const float* egamma = (const float*)d_in[7];
    const float* ebeta  = (const float*)d_in[8];
    float*       out    = (float*)d_out;

    // Workspace layout: xn f16 table (12.8 MB) then WT f16 (192 KB).
    _Float16* xn = (_Float16*)d_ws;
    _Float16* WT = (_Float16*)((char*)d_ws + (size_t)N_NODES * D * sizeof(_Float16));

    node_ln_kernel<<<N_NODES / 8, 256, 0, stream>>>(x, ngamma, nbeta, xn);
    wt_kernel<<<(ODIM * KDIM) / 256, 256, 0, stream>>>(W, WT);
    edge_kernel<<<N_EDGES / 32, 256, 0, stream>>>(eidx, ea, xn, WT, b,
                                                  egamma, ebeta, out);
}
